// MedicalGraphEncoder_17635135718023
// MI455X (gfx1250) — matrix operand order repaired and checked by reading, not compiled, not run on hardware
//
#include <hip/hip_runtime.h>
#include <cstdint>
#include <cstddef>

#define D_FEAT 768
#define H1 4
#define HC1 (H1 * D_FEAT) /* 3072 */
#define NEG_SLOPE 0.2f

typedef __attribute__((ext_vector_type(16))) __bf16 v16bf;
typedef __attribute__((ext_vector_type(8)))  float  v8f;
typedef int i32x4 __attribute__((vector_size(16)));

union Frag16 { v16bf v; unsigned int u[8]; };

// CDNA5 async-to-LDS availability (guarded: falls back to manual staging)
#if defined(__has_builtin)
#  if __has_builtin(__builtin_amdgcn_global_load_async_to_lds_b128)
#    define HAVE_ASYNC_B128 1
#  endif
#  if __has_builtin(__builtin_amdgcn_s_wait_asynccnt)
#    define HAVE_WAIT_ASYNC 1
#  endif
#endif

// flat->AS1 / flat->AS3 via integer round trip (flat global == AS1 address;
// flat LDS low 32 bits == LDS offset per CDNA5 aperture mapping).
// Builtin wants typed b128 (int4) pointers per the clang signature.
#define AS1P(p) ((__attribute__((address_space(1))) i32x4*)(uintptr_t)(p))
#define AS3P(p) ((__attribute__((address_space(3))) i32x4*)(uintptr_t)(p))

__device__ __forceinline__ unsigned short f2bf(float f) {
  unsigned int u = __float_as_uint(f);
  u += 0x7FFFu + ((u >> 16) & 1u);   // round-to-nearest-even
  return (unsigned short)(u >> 16);
}

// ---------------------------------------------------------------- fills / cvt
__global__ void fill_f32_kernel(float* p, float v, long n) {
  long i = (long)blockIdx.x * blockDim.x + threadIdx.x;
  if (i < n) p[i] = v;
}
__global__ void fill_i32_kernel(int* p, int v, long n) {
  long i = (long)blockIdx.x * blockDim.x + threadIdx.x;
  if (i < n) p[i] = v;
}
__global__ void cvt_f32_bf16_kernel(const float* __restrict__ in,
                                    unsigned short* __restrict__ out, long n) {
  long i = (long)blockIdx.x * blockDim.x + threadIdx.x;
  if (i < n) out[i] = f2bf(in[i]);
}

// ---------------------------------------------------------------- WMMA GEMM
// C[M,N] = A[M,K](bf16) * B[K,N](bf16) + bias[N], f32 accumulate.
// Block = 256 threads = 8 waves. Block tile 256(M) x 64(N), BK = 32.
// Wave w computes rows [32w,32w+32) x 64 cols = 8 wmma tiles (B reused 2x).
#define BM 256
#define BN 64
#define BK 32
#define APAD 20  /* dwords per A row: 80B stride = 16B aligned + bank-free */

__global__ __launch_bounds__(256) void gemm_bf16_wmma(
    const unsigned short* __restrict__ A,   // [M,K] bf16 row-major
    const unsigned short* __restrict__ B,   // [K,N] bf16 row-major
    const float* __restrict__ bias,         // [N]
    float* __restrict__ C,                  // [M,N] f32
    int M, int K, int N) {
  __shared__ unsigned int   ldsA[BM][APAD];      // 20 KB
  __shared__ unsigned short ldsBt[BN][BK + 2];   // transposed B, 68B row stride

  const int tid  = threadIdx.x;
  const int wave = tid >> 5;    // 0..7
  const int lane = tid & 31;
  const int half = lane >> 4;   // K-half selector per ISA layout
  const int l16  = lane & 15;

  const int rowBase = blockIdx.y * BM;
  const int colBase = blockIdx.x * BN;

  v8f accLo[4], accHi[4];
#pragma unroll
  for (int j = 0; j < 4; ++j)
#pragma unroll
    for (int r = 0; r < 8; ++r) { accLo[j][r] = 0.0f; accHi[j][r] = 0.0f; }

  for (int k0 = 0; k0 < K; k0 += BK) {
    // ---- stage A tile 256x32 bf16 (4 x b128 per thread) ----
#pragma unroll
    for (int it = 0; it < 4; ++it) {
      int idx  = tid + it * 256;        // 0..1023 uint4 slots
      int r    = idx >> 2;              // row in tile
      int c4   = idx & 3;               // uint4 within row
      int grow = rowBase + r;
#ifdef HAVE_ASYNC_B128
      if (grow < M) {
        // CDNA5 async DMA into LDS (tracked by ASYNCcnt, no VGPR data)
        __builtin_amdgcn_global_load_async_to_lds_b128(
            AS1P(A + (size_t)grow * K + k0 + c4 * 8),
            AS3P(&ldsA[r][c4 * 4]), 0, 0);
      } else {
        uint4 z = make_uint4(0u, 0u, 0u, 0u);
        *(uint4*)&ldsA[r][c4 * 4] = z;
      }
#else
      uint4 val = make_uint4(0u, 0u, 0u, 0u);
      if (grow < M) {
        const uint4* p = (const uint4*)(A + (size_t)grow * K + k0);
        val = p[c4];
        if (k0 + BK < K)  // prefetch next K tile -> global_prefetch_b8
          __builtin_prefetch((const void*)(A + (size_t)grow * K + k0 + BK), 0, 1);
      }
      *(uint4*)&ldsA[r][c4 * 4] = val;
#endif
    }
    // ---- stage B tile 32x64 bf16, transposed into LDS (1 x b128/thread) ----
    {
      int r  = tid >> 3;                // k row 0..31
      int c8 = tid & 7;                 // group of 8 cols
      const uint4* p = (const uint4*)(B + (size_t)(k0 + r) * N + colBase + c8 * 8);
      uint4 v = *p;
      unsigned short s[8];
      s[0] = (unsigned short)(v.x & 0xFFFFu); s[1] = (unsigned short)(v.x >> 16);
      s[2] = (unsigned short)(v.y & 0xFFFFu); s[3] = (unsigned short)(v.y >> 16);
      s[4] = (unsigned short)(v.z & 0xFFFFu); s[5] = (unsigned short)(v.z >> 16);
      s[6] = (unsigned short)(v.w & 0xFFFFu); s[7] = (unsigned short)(v.w >> 16);
#pragma unroll
      for (int i = 0; i < 8; ++i) ldsBt[c8 * 8 + i][r] = s[i];
    }
#ifdef HAVE_ASYNC_B128
#  ifdef HAVE_WAIT_ASYNC
    __builtin_amdgcn_s_wait_asynccnt(0);
#  else
    asm volatile("s_wait_asynccnt 0" ::: "memory");
#  endif
#endif
    __syncthreads();

    // ---- A fragments: ISA 16-bit A 16x32 layout (K pairs interleaved) ----
    Frag16 a0, a1;
    {
      int m0 = wave * 32 + l16;
#pragma unroll
      for (int p = 0; p < 8; ++p) {
        int kp = (p < 4) ? (p + 4 * half) : (p + 4 + 4 * half);
        a0.u[p] = ldsA[m0][kp];
        a1.u[p] = ldsA[m0 + 16][kp];
      }
    }
    // ---- 8 WMMA per K-step; each B fragment reused for 2 row tiles ----
#pragma unroll
    for (int j = 0; j < 4; ++j) {
      Frag16 bfr;
      int n = j * 16 + l16;
#pragma unroll
      for (int p = 0; p < 8; ++p) {
        int kp = (p < 4) ? (p + 4 * half) : (p + 4 + 4 * half);   // B: lanes 0-15 K=0..7,16..23, lanes 16-31 K=8..15,24..31
        bfr.u[p] = *(const unsigned int*)&ldsBt[n][kp * 2];
      }
      accLo[j] = __builtin_amdgcn_wmma_f32_16x16x32_bf16(
          false, a0.v, false, bfr.v, (short)0, accLo[j], false, false);
      accHi[j] = __builtin_amdgcn_wmma_f32_16x16x32_bf16(
          false, a1.v, false, bfr.v, (short)0, accHi[j], false, false);
    }
    __syncthreads();
  }

  // ---- epilogue: D layout VGPR r -> M = r + 8*half, N = lane&15 ----
#pragma unroll
  for (int j = 0; j < 4; ++j) {
    int n  = colBase + j * 16 + l16;
    float bv = bias[n];
#pragma unroll
    for (int r = 0; r < 8; ++r) {
      int gm0 = rowBase + wave * 32 + r + 8 * half;
      int gm1 = gm0 + 16;
      if (gm0 < M) C[(size_t)gm0 * N + n] = accLo[j][r] + bv;
      if (gm1 < M) C[(size_t)gm1 * N + n] = accHi[j][r] + bv;
    }
  }
}

// ---------------------------------------------------------------- attention
// One wave per (edge, head): leaky_relu(xi+xj) . att, then atomicMax (ordered
// int encoding of float) into per-destination running max.
__global__ __launch_bounds__(256) void edge_logits_kernel(
    const float* __restrict__ xl, const float* __restrict__ xr,
    const float* __restrict__ att, const int* __restrict__ ei,
    int E, int Etot, int H, int C,
    float* __restrict__ elog, int* __restrict__ mord) {
  int gw   = (int)((blockIdx.x * (long)blockDim.x + threadIdx.x) >> 5);
  int lane = threadIdx.x & 31;
  if (gw >= Etot * H) return;
  int e = gw / H, h = gw - e * H;
  int s = (e < E) ? ei[e] : (e - E);
  int d = (e < E) ? ei[E + e] : (e - E);
  const float* pj = xl + ((size_t)s * H + h) * C;
  const float* pi = xr + ((size_t)d * H + h) * C;
  const float* pa = att + (size_t)h * C;
  float acc = 0.0f;
  for (int c = lane; c < C; c += 32) {
    float v = pi[c] + pj[c];
    v = (v > 0.0f) ? v : NEG_SLOPE * v;
    acc += pa[c] * v;
  }
#pragma unroll
  for (int off = 16; off > 0; off >>= 1) acc += __shfl_xor(acc, off, 32);
  if (lane == 0) {
    elog[(size_t)e * H + h] = acc;
    int i   = __float_as_int(acc);
    int ord = (i >= 0) ? i : (i ^ 0x7FFFFFFF);   // monotone float->int
    atomicMax(&mord[d * H + h], ord);
  }
}

__global__ void edge_exp_kernel(const int* __restrict__ ei, int E, int Etot, int H,
                                const int* __restrict__ mord,
                                float* __restrict__ elog,
                                float* __restrict__ denom) {
  int idx = blockIdx.x * blockDim.x + threadIdx.x;
  if (idx >= Etot * H) return;
  int e = idx / H, h = idx - e * H;
  int d  = (e < E) ? ei[E + e] : (e - E);
  int mo = mord[d * H + h];
  int mi = (mo >= 0) ? mo : (mo ^ 0x7FFFFFFF);
  float m = __int_as_float(mi);
  float w = __expf(elog[idx] - m);
  elog[idx] = w;
  atomicAdd(&denom[d * H + h], w);
}

// One block per edge; coalesced row read + coalesced atomic f32 scatter-add.
__global__ __launch_bounds__(256) void edge_aggregate_kernel(
    const int* __restrict__ ei, int E, int Etot, int H, int C,
    const float* __restrict__ xl, const float* __restrict__ elog,
    const float* __restrict__ denom, float* __restrict__ agg) {
  int e = blockIdx.x;
  if (e >= Etot) return;
  int s = (e < E) ? ei[e] : (e - E);
  int d = (e < E) ? ei[E + e] : (e - E);
  for (int h = 0; h < H; ++h) {
    float alpha = elog[(size_t)e * H + h] / (denom[d * H + h] + 1e-16f);
    const float* src = xl + ((size_t)s * H + h) * C;
    float*       dst = agg + ((size_t)d * H + h) * C;
    for (int c = threadIdx.x; c < C; c += blockDim.x)
      atomicAdd(&dst[c], alpha * src[c]);
  }
}

// layer1 epilogue: +bias, relu, convert to bf16 for layer-2 WMMA GEMMs
__global__ void bias_relu_bf16_kernel(const float* __restrict__ agg,
                                      const float* __restrict__ bias,
                                      unsigned short* __restrict__ out,
                                      long n, int F) {
  long i = (long)blockIdx.x * blockDim.x + threadIdx.x;
  if (i < n) {
    float v = agg[i] + bias[(int)(i % F)];
    out[i] = f2bf(v > 0.0f ? v : 0.0f);
  }
}

// final: first N_PATHOLOGY rows, + bias2 (H=1 so mean over heads is identity)
__global__ void final_out_kernel(const float* __restrict__ agg,
                                 const float* __restrict__ bias,
                                 float* __restrict__ out, int n, int F) {
  int i = blockIdx.x * blockDim.x + threadIdx.x;
  if (i < n) out[i] = agg[i] + bias[i % F];
}

// ---------------------------------------------------------------- launch
extern "C" void kernel_launch(void* const* d_in, const int* in_sizes, int n_in,
                              void* d_out, int out_size, void* d_ws, size_t ws_size,
                              hipStream_t stream) {
  const float* x     = (const float*)d_in[0];
  const int*   ei    = (const int*)  d_in[1];
  const float* Wl1   = (const float*)d_in[2];
  const float* bl1   = (const float*)d_in[3];
  const float* Wr1   = (const float*)d_in[4];
  const float* br1   = (const float*)d_in[5];
  const float* att1  = (const float*)d_in[6];
  const float* bias1 = (const float*)d_in[7];
  const float* Wl2   = (const float*)d_in[8];
  const float* bl2   = (const float*)d_in[9];
  const float* Wr2   = (const float*)d_in[10];
  const float* br2   = (const float*)d_in[11];
  const float* att2  = (const float*)d_in[12];
  const float* bias2 = (const float*)d_in[13];

  const int Nn   = in_sizes[0] / D_FEAT;  // 10000 nodes
  const int E    = in_sizes[1] / 2;       // 50000 edges
  const int Etot = E + Nn;                // + self loops

  // ---- workspace carve-out (256B aligned); layer-2 aliases dead layer-1 bufs
  char*  base = (char*)d_ws;
  size_t off  = 0;
  auto take = [&](size_t bytes) -> char* {
    char* p = base + off;
    off += bytes;
    off = (off + 255) & ~(size_t)255;
    return p;
  };
  unsigned short* xbf = (unsigned short*)take((size_t)Nn * D_FEAT * 2);
  unsigned short* w1l = (unsigned short*)take((size_t)D_FEAT * HC1 * 2);
  unsigned short* w1r = (unsigned short*)take((size_t)D_FEAT * HC1 * 2);
  unsigned short* w2l = (unsigned short*)take((size_t)HC1 * D_FEAT * 2);
  unsigned short* w2r = (unsigned short*)take((size_t)HC1 * D_FEAT * 2);
  unsigned short* hbf = (unsigned short*)take((size_t)Nn * HC1 * 2);
  float* ebuf  = (float*)take((size_t)Etot * H1 * 4);
  int*   mord  = (int*)  take((size_t)Nn * H1 * 4);
  float* denom = (float*)take((size_t)Nn * H1 * 4);
  float* xl1   = (float*)take((size_t)Nn * HC1 * 4);
  float* xr1   = (float*)take((size_t)Nn * HC1 * 4);
  float* agg1  = (float*)take((size_t)Nn * HC1 * 4);
  float* xl2  = xl1;    // dead after edge_aggregate (layer 1)
  float* xr2  = xr1;
  float* agg2 = agg1;   // dead after bias_relu_bf16

  const int TB = 256;
  auto blk = [](long n, int t) { return (unsigned)((n + t - 1) / t); };

  // ---- precision conversion (f32 -> bf16) for WMMA operands
  cvt_f32_bf16_kernel<<<blk((long)Nn * D_FEAT, TB), TB, 0, stream>>>(x, xbf, (long)Nn * D_FEAT);
  cvt_f32_bf16_kernel<<<blk((long)D_FEAT * HC1, TB), TB, 0, stream>>>(Wl1, w1l, (long)D_FEAT * HC1);
  cvt_f32_bf16_kernel<<<blk((long)D_FEAT * HC1, TB), TB, 0, stream>>>(Wr1, w1r, (long)D_FEAT * HC1);
  cvt_f32_bf16_kernel<<<blk((long)HC1 * D_FEAT, TB), TB, 0, stream>>>(Wl2, w2l, (long)HC1 * D_FEAT);
  cvt_f32_bf16_kernel<<<blk((long)HC1 * D_FEAT, TB), TB, 0, stream>>>(Wr2, w2r, (long)HC1 * D_FEAT);

  // ================= layer 1 =================
  fill_i32_kernel<<<blk((long)Nn * H1, TB), TB, 0, stream>>>(mord, (int)0x80000000, (long)Nn * H1);
  fill_f32_kernel<<<blk((long)Nn * H1, TB), TB, 0, stream>>>(denom, 0.0f, (long)Nn * H1);
  fill_f32_kernel<<<blk((long)Nn * HC1, TB), TB, 0, stream>>>(agg1, 0.0f, (long)Nn * HC1);

  dim3 g1(HC1 / BN, (Nn + BM - 1) / BM);
  gemm_bf16_wmma<<<g1, 256, 0, stream>>>(xbf, w1l, bl1, xl1, Nn, D_FEAT, HC1);
  gemm_bf16_wmma<<<g1, 256, 0, stream>>>(xbf, w1r, br1, xr1, Nn, D_FEAT, HC1);

  long waves1 = (long)Etot * H1;
  edge_logits_kernel<<<blk(waves1 * 32, TB), TB, 0, stream>>>(
      xl1, xr1, att1, ei, E, Etot, H1, D_FEAT, ebuf, mord);
  edge_exp_kernel<<<blk(waves1, TB), TB, 0, stream>>>(ei, E, Etot, H1, mord, ebuf, denom);
  edge_aggregate_kernel<<<Etot, TB, 0, stream>>>(ei, E, Etot, H1, D_FEAT, xl1, ebuf, denom, agg1);
  bias_relu_bf16_kernel<<<blk((long)Nn * HC1, TB), TB, 0, stream>>>(
      agg1, bias1, hbf, (long)Nn * HC1, HC1);

  // ================= layer 2 (H = 1) =================
  fill_i32_kernel<<<blk((long)Nn, TB), TB, 0, stream>>>(mord, (int)0x80000000, (long)Nn);
  fill_f32_kernel<<<blk((long)Nn, TB), TB, 0, stream>>>(denom, 0.0f, (long)Nn);
  fill_f32_kernel<<<blk((long)Nn * D_FEAT, TB), TB, 0, stream>>>(agg2, 0.0f, (long)Nn * D_FEAT);

  dim3 g2(D_FEAT / BN, (Nn + BM - 1) / BM);
  gemm_bf16_wmma<<<g2, 256, 0, stream>>>(hbf, w2l, bl2, xl2, Nn, HC1, D_FEAT);
  gemm_bf16_wmma<<<g2, 256, 0, stream>>>(hbf, w2r, br2, xr2, Nn, HC1, D_FEAT);

  long waves2 = (long)Etot;
  edge_logits_kernel<<<blk(waves2 * 32, TB), TB, 0, stream>>>(
      xl2, xr2, att2, ei, E, Etot, 1, D_FEAT, ebuf, mord);
  edge_exp_kernel<<<blk(waves2, TB), TB, 0, stream>>>(ei, E, Etot, 1, mord, ebuf, denom);
  edge_aggregate_kernel<<<Etot, TB, 0, stream>>>(ei, E, Etot, 1, D_FEAT, xl2, ebuf, denom, agg2);

  final_out_kernel<<<blk(out_size, TB), TB, 0, stream>>>(
      agg2, bias2, (float*)d_out, out_size, D_FEAT);
}
